// SwinTransformerBlock_48232482734196
// MI455X (gfx1250) — matrix operand order repaired and checked by reading, not compiled, not run on hardware
//
#include <hip/hip_runtime.h>

typedef _Float16 half_t;
typedef __attribute__((ext_vector_type(16))) _Float16 v16h;
typedef __attribute__((ext_vector_type(8)))  float    v8f;

#define ROWS 50176   // 16 * 8 * 8 * 49  == B*L token rows (window order)
#define CDIM 192
#define KDIM 192
#define HIDD 768
#define MLPW 2

union Frag16 { v16h v; uint4 q[2]; };

// A-matrix fragment (16x32 f16, row-major source, lda in halfs).
// Lanes 0-15: elems 0..7 = K k0..k0+7, elems 8..15 = K k0+16..k0+23.
// Lanes 16-31: +8 on the K base (K k0+8.. / k0+24..).
__device__ __forceinline__ v16h load_a_frag(const half_t* A, int lda, int row, int k0, int lane) {
  const half_t* p = A + (size_t)row * lda + k0 + ((lane & 16) ? 8 : 0);
  Frag16 f;
  f.q[0] = *reinterpret_cast<const uint4*>(p);
  f.q[1] = *reinterpret_cast<const uint4*>(p + 16);
  return f.v;
}

// B-matrix fragment (32x16 f16). Source is "WT" layout: column-major weights,
// i.e. WT[col*ldw + k] with K contiguous. Lanes 0-15: elems 0..7 = K k0..k0+7,
// elems 8..15 = K k0+16..k0+23; lanes 16-31: +8 on the K base.
__device__ __forceinline__ v16h load_b_frag(const half_t* WT, int ldw, int col, int k0, int lane) {
  const half_t* p = WT + (size_t)col * ldw + k0 + ((lane & 16) ? 8 : 0);
  Frag16 f;
  f.q[0] = *reinterpret_cast<const uint4*>(p);
  f.q[1] = *reinterpret_cast<const uint4*>(p + 16);
  return f.v;
}

__device__ __forceinline__ v8f wmma32(v16h a, v16h b, v8f c) {
  return __builtin_amdgcn_wmma_f32_16x16x32_f16(false, a, false, b, (short)0, c, false, false);
}

__device__ __forceinline__ v8f vzero8() {
  v8f z = {0.f, 0.f, 0.f, 0.f, 0.f, 0.f, 0.f, 0.f};
  return z;
}

__device__ __forceinline__ float gelu_exact(float x) {
  return 0.5f * x * (1.0f + erff(x * 0.70710678118654752f));
}

// ---------------------------------------------------------------------------
// Weight prep: fp32 W (KxN row-major) -> f16 WT (N-major, K contiguous)
// ---------------------------------------------------------------------------
__global__ __launch_bounds__(256) void wt_f16(const float* __restrict__ W,
                                              half_t* __restrict__ WT, int K, int N) {
  int idx = blockIdx.x * 256 + threadIdx.x;
  if (idx >= K * N) return;
  int k = idx / N, n = idx % N;
  WT[(size_t)n * K + k] = (half_t)W[idx];
}

// ---------------------------------------------------------------------------
// LN(se), LN(de), pass-through(co) + cyclic shift (-3,-3) + window partition.
// One wave per output (window-order) token row. 6 channels per lane.
// ---------------------------------------------------------------------------
__global__ __launch_bounds__(256) void ln_shift_part(
    const float* __restrict__ se, const float* __restrict__ de, const float* __restrict__ co,
    const float* __restrict__ g1, const float* __restrict__ b1,
    const float* __restrict__ g2, const float* __restrict__ b2,
    half_t* __restrict__ xse, half_t* __restrict__ xde, half_t* __restrict__ xco) {
  int wave = threadIdx.x >> 5, lane = threadIdx.x & 31;
  int r = blockIdx.x * 8 + wave;            // 0..ROWS-1 (window order)
  int w = r / 49, tok = r % 49;
  int b = w >> 6, wy = (w >> 3) & 7, wx = w & 7;
  int ty = tok / 7, tx = tok % 7;
  int hs = wy * 7 + ty + 3; if (hs >= 56) hs -= 56;     // roll by -3
  int wc = wx * 7 + tx + 3; if (wc >= 56) wc -= 56;
  size_t src = ((size_t)b * 3136 + hs * 56 + wc) * CDIM;
  float xs[6], xd[6];
  float s1 = 0.f, s2 = 0.f, t1 = 0.f, t2 = 0.f;
#pragma unroll
  for (int i = 0; i < 6; i++) {
    float v = se[src + lane * 6 + i]; xs[i] = v; s1 += v; s2 += v * v;
    float u = de[src + lane * 6 + i]; xd[i] = u; t1 += u; t2 += u * u;
  }
#pragma unroll
  for (int off = 16; off > 0; off >>= 1) {
    s1 += __shfl_xor(s1, off); s2 += __shfl_xor(s2, off);
    t1 += __shfl_xor(t1, off); t2 += __shfl_xor(t2, off);
  }
  float ms = s1 * (1.f / 192.f), vs = s2 * (1.f / 192.f) - ms * ms;
  float md = t1 * (1.f / 192.f), vd = t2 * (1.f / 192.f) - md * md;
  float rs = rsqrtf(vs + 1e-5f), rd = rsqrtf(vd + 1e-5f);
  size_t dst = (size_t)r * CDIM;
#pragma unroll
  for (int i = 0; i < 6; i++) {
    int c = lane * 6 + i;
    xse[dst + c] = (half_t)((xs[i] - ms) * rs * g1[c] + b1[c]);
    xde[dst + c] = (half_t)((xd[i] - md) * rd * g2[c] + b2[c]);
    xco[dst + c] = (half_t)co[src + c];
  }
}

// ---------------------------------------------------------------------------
// LayerNorm of fp32 rows -> f16 rows (one wave per row).
// ---------------------------------------------------------------------------
__global__ __launch_bounds__(256) void ln_rows(
    const float* __restrict__ x, const float* __restrict__ g, const float* __restrict__ b,
    half_t* __restrict__ y) {
  int wave = threadIdx.x >> 5, lane = threadIdx.x & 31;
  int r = blockIdx.x * 8 + wave;
  size_t base = (size_t)r * CDIM;
  float v[6]; float s1 = 0.f, s2 = 0.f;
#pragma unroll
  for (int i = 0; i < 6; i++) { float t = x[base + lane * 6 + i]; v[i] = t; s1 += t; s2 += t * t; }
#pragma unroll
  for (int off = 16; off > 0; off >>= 1) { s1 += __shfl_xor(s1, off); s2 += __shfl_xor(s2, off); }
  float m = s1 * (1.f / 192.f), var = s2 * (1.f / 192.f) - m * m;
  float rs = rsqrtf(var + 1e-5f);
#pragma unroll
  for (int i = 0; i < 6; i++) {
    int c = lane * 6 + i;
    y[base + c] = (half_t)((v[i] - m) * rs * g[c] + b[c]);
  }
}

// ---------------------------------------------------------------------------
// WMMA GEMM, K fixed to 192 per A operand, fully unrolled.
// out[M x N] = A1[M x 192] @ WT-slice(koff 0) (+ A2 @ WT-slice(koff 192)) + bias.
// WT is N-major f16 (column n at WT + n*LDW, K contiguous). The 64-column
// weight tile is staged in LDS once per block and shared by all 8 waves.
// All 6 A-fragments are preloaded in one clause so loads overlap the WMMAs.
// ---------------------------------------------------------------------------
template <int LDW, bool HAS_A2, bool OUT_F16>
__global__ __launch_bounds__(256) void gemm_f16w(
    const half_t* __restrict__ A1, const half_t* __restrict__ A2,
    const half_t* __restrict__ WT, const float* __restrict__ bias,
    void* __restrict__ outp, int N) {
  __shared__ __align__(16) half_t Bs[64 * LDW];
  int tid = threadIdx.x;
  int wave = tid >> 5, lane = tid & 31;
  int lr = lane & 15, hi = (lane >> 4) & 1;
  int row0 = (blockIdx.x * 8 + wave) * 16;
  int nb = blockIdx.y * 64;

  // Cooperative stage of the 64 x LDW weight tile (uint4 = 8 halfs per chunk).
  {
    constexpr int CPL = LDW / 8;            // chunks per column
    constexpr int CHUNKS = 64 * CPL;
#pragma unroll
    for (int i = tid; i < CHUNKS; i += 256) {
      int col = i / CPL, kc = i - col * CPL;
      *reinterpret_cast<uint4*>(&Bs[col * LDW + kc * 8]) =
          *reinterpret_cast<const uint4*>(WT + (size_t)(nb + col) * LDW + kc * 8);
    }
  }
  __syncthreads();

  v8f acc[4];
#pragma unroll
  for (int j = 0; j < 4; j++) acc[j] = vzero8();

  v16h af[6];
#pragma unroll
  for (int kk = 0; kk < 6; kk++) af[kk] = load_a_frag(A1, KDIM, row0 + lr, kk * 32, lane);
#pragma unroll
  for (int kk = 0; kk < 6; kk++) {
#pragma unroll
    for (int j = 0; j < 4; j++)
      acc[j] = wmma32(af[kk], load_b_frag(Bs, LDW, 16 * j + lr, kk * 32, lane), acc[j]);
  }
  if (HAS_A2) {
#pragma unroll
    for (int kk = 0; kk < 6; kk++) af[kk] = load_a_frag(A2, KDIM, row0 + lr, kk * 32, lane);
#pragma unroll
    for (int kk = 0; kk < 6; kk++) {
#pragma unroll
      for (int j = 0; j < 4; j++)
        acc[j] = wmma32(af[kk], load_b_frag(Bs, LDW, 16 * j + lr, 192 + kk * 32, lane), acc[j]);
    }
  }

#pragma unroll
  for (int j = 0; j < 4; j++) {
    int n = nb + 16 * j + lr;
    float bv = bias[n];
#pragma unroll
    for (int v = 0; v < 8; v++) {
      int m = row0 + v + hi * 8;
      float x = acc[j][v] + bv;
      if (OUT_F16) ((half_t*)outp)[(size_t)m * N + n] = (half_t)x;
      else         ((float*)outp)[(size_t)m * N + n] = x;
    }
  }
}

// ---------------------------------------------------------------------------
// Per-(window, head) attention. 1 wave, 40KB LDS.
// S = Q K^T (WMMA), softmax(S*scale + rel_pos_bias + shift_mask), O = P V.
// B-fragments are invariant across the M-tile loops -> hoisted into registers.
// ---------------------------------------------------------------------------
__global__ __launch_bounds__(32) void attn_win(
    const half_t* __restrict__ qk, const half_t* __restrict__ vse, const half_t* __restrict__ vde,
    const float* __restrict__ rpb, const float* __restrict__ maskm,
    half_t* __restrict__ ose, half_t* __restrict__ ode) {
  __shared__ float S[64 * 64];
  __shared__ __align__(16) half_t P[64 * 64];
  __shared__ __align__(16) half_t qs[64 * 32];
  __shared__ __align__(16) half_t ks[64 * 32];
  __shared__ __align__(16) half_t vts[32 * 64];   // V_se^T : [chan][token]
  __shared__ __align__(16) half_t vtd[32 * 64];   // V_de^T
  int w = blockIdx.x, h = blockIdx.y;
  int lane = threadIdx.x, lr = lane & 15, hi = (lane >> 4) & 1;
  size_t rbase = (size_t)w * 49;

  // Stage zero-padded Q, K (64 tokens x 32 chans) and transposed V's.
  for (int i = lane; i < 64 * 32; i += 32) {
    int t = i >> 5, c = i & 31;
    half_t qv = (half_t)0.f, kv = (half_t)0.f;
    if (t < 49) {
      qv = qk[(rbase + t) * 384 + h * 32 + c];
      kv = qk[(rbase + t) * 384 + 192 + h * 32 + c];
    }
    qs[i] = qv; ks[i] = kv;
  }
  for (int i = lane; i < 32 * 64; i += 32) {
    int c = i >> 6, t = i & 63;
    half_t a = (half_t)0.f, bv = (half_t)0.f;
    if (t < 49) {
      a  = vse[(rbase + t) * CDIM + h * 32 + c];
      bv = vde[(rbase + t) * CDIM + h * 32 + c];
    }
    vts[i] = a; vtd[i] = bv;
  }
  __syncthreads();

  // S = Q @ K^T : hoist the 4 K^T fragments, then 4x4 tiles (one wmma each).
  {
    v16h bk[4];
#pragma unroll
    for (int nt = 0; nt < 4; nt++) bk[nt] = load_b_frag(ks, 32, nt * 16 + lr, 0, lane);
#pragma unroll
    for (int mt = 0; mt < 4; mt++) {
      v16h aq = load_a_frag(qs, 32, mt * 16 + lr, 0, lane);
#pragma unroll
      for (int nt = 0; nt < 4; nt++) {
        v8f acc = vzero8();
        acc = wmma32(aq, bk[nt], acc);
#pragma unroll
        for (int v = 0; v < 8; v++)
          S[(mt * 16 + v + hi * 8) * 64 + nt * 16 + lr] = acc[v];
      }
    }
  }
  __syncthreads();

  // Softmax rows (scale folded here; bias + mask added before softmax).
  const float scale = 0.17677669529663687f;   // 32^-0.5
  const float* mk = maskm + (size_t)(w & 63) * 49 * 49;
  for (int r = lane; r < 64; r += 32) {
    if (r < 49) {
      int yi = r / 7, xi = r % 7;
      float mx = -1e30f;
      for (int j = 0; j < 49; j++) {
        int yj = j / 7, xj = j % 7;
        int ridx = (yi - yj + 6) * 13 + (xi - xj + 6);
        float lg = S[r * 64 + j] * scale + rpb[ridx * 6 + h] + mk[r * 49 + j];
        S[r * 64 + j] = lg;
        mx = fmaxf(mx, lg);
      }
      float sum = 0.f;
      for (int j = 0; j < 49; j++) { float e = expf(S[r * 64 + j] - mx); S[r * 64 + j] = e; sum += e; }
      float inv = 1.0f / sum;
      for (int j = 0; j < 64; j++)
        P[r * 64 + j] = (j < 49) ? (half_t)(S[r * 64 + j] * inv) : (half_t)0.f;
    } else {
      for (int j = 0; j < 64; j++) P[r * 64 + j] = (half_t)0.f;
    }
  }
  __syncthreads();

  // O = P @ V  (K padded to 64, two 32-steps; padded P cols are zero).
  // Hoist the 8 V fragments (2 nt x 2 ksteps x {se,de}).
  {
    v16h bs[2][2], bd[2][2];
#pragma unroll
    for (int nt = 0; nt < 2; nt++) {
      bs[nt][0] = load_b_frag(vts, 64, nt * 16 + lr, 0, lane);
      bs[nt][1] = load_b_frag(vts, 64, nt * 16 + lr, 32, lane);
      bd[nt][0] = load_b_frag(vtd, 64, nt * 16 + lr, 0, lane);
      bd[nt][1] = load_b_frag(vtd, 64, nt * 16 + lr, 32, lane);
    }
#pragma unroll
    for (int mt = 0; mt < 4; mt++) {
      v16h a0 = load_a_frag(P, 64, mt * 16 + lr, 0, lane);
      v16h a1 = load_a_frag(P, 64, mt * 16 + lr, 32, lane);
#pragma unroll
      for (int nt = 0; nt < 2; nt++) {
        v8f as = vzero8(), ad = vzero8();
        as = wmma32(a0, bs[nt][0], as);
        as = wmma32(a1, bs[nt][1], as);
        ad = wmma32(a0, bd[nt][0], ad);
        ad = wmma32(a1, bd[nt][1], ad);
#pragma unroll
        for (int v = 0; v < 8; v++) {
          int m = mt * 16 + v + hi * 8;
          if (m < 49) {
            int n = h * 32 + nt * 16 + lr;
            ose[(rbase + m) * CDIM + n] = (half_t)as[v];
            ode[(rbase + m) * CDIM + n] = (half_t)ad[v];
          }
        }
      }
    }
  }
}

// ---------------------------------------------------------------------------
// Window reverse + roll(+3,+3) + residual add into d_out (both streams).
// ---------------------------------------------------------------------------
__global__ __launch_bounds__(256) void scatter_residual(
    const float* __restrict__ se, const float* __restrict__ de,
    const float* __restrict__ pse, const float* __restrict__ pde,
    float* __restrict__ ose, float* __restrict__ ode) {
  int idx = blockIdx.x * 256 + threadIdx.x;          // < ROWS*CDIM exactly
  int c = idx % CDIM;
  int t = idx / CDIM;
  int l = t % 3136, b = t / 3136;
  int hh = l / 56, ww = l % 56;
  int hs = hh - 3; if (hs < 0) hs += 56;             // inverse of roll(+3)
  int wsh = ww - 3; if (wsh < 0) wsh += 56;
  int wy = hs / 7, ty = hs % 7, wx = wsh / 7, tx = wsh % 7;
  size_t row = ((size_t)b * 64 + wy * 8 + wx) * 49 + ty * 7 + tx;
  ose[idx] = se[idx] + pse[row * CDIM + c];
  ode[idx] = de[idx] + pde[row * CDIM + c];
}

// ---------------------------------------------------------------------------
// Fused MLP: y = x + (gelu(ln2(x) @ W1 + b1) @ W2 + b2), residual read from
// `out` in place. One wave per 16-row tile; 16x768 f16 hidden tile in LDS.
// ---------------------------------------------------------------------------
__global__ __launch_bounds__(64) void mlp_fused(
    const half_t* __restrict__ X, const half_t* __restrict__ W1T, const float* __restrict__ b1,
    const half_t* __restrict__ W2T, const float* __restrict__ b2, float* __restrict__ out) {
  __shared__ __align__(16) half_t hb_all[MLPW * 16 * HIDD];   // 48KB
  int wave = threadIdx.x >> 5, lane = threadIdx.x & 31;
  int lr = lane & 15, hi = (lane >> 4) & 1;
  int row0 = (blockIdx.x * MLPW + wave) * 16;
  half_t* hb = hb_all + wave * 16 * HIDD;

  // Cache the 6 A-fragments of this 16x192 tile.
  v16h af[6];
#pragma unroll
  for (int kk = 0; kk < 6; kk++) af[kk] = load_a_frag(X, CDIM, row0 + lr, kk * 32, lane);

  // GEMM1 (192 -> 768) + exact GELU into LDS.
  for (int nt = 0; nt < 48; nt++) {
    __builtin_prefetch(W1T + (size_t)(nt * 16 + 16 + lr) * CDIM, 0, 1);  // global_prefetch_b8
    v8f acc = vzero8();
#pragma unroll
    for (int kk = 0; kk < 6; kk++)
      acc = wmma32(af[kk], load_b_frag(W1T, CDIM, nt * 16 + lr, kk * 32, lane), acc);
    int n = nt * 16 + lr;
    float bv = b1[n];
#pragma unroll
    for (int v = 0; v < 8; v++)
      hb[(v + hi * 8) * HIDD + n] = (half_t)gelu_exact(acc[v] + bv);
  }
  // Per-wave LDS region, DS ops are in-order per wave: no barrier needed.

  // GEMM2 (768 -> 192), 12 persistent accumulators, A double-buffered from LDS
  // with the last k-step peeled (straight-line, no conditionals).
  v8f oacc[12];
#pragma unroll
  for (int j = 0; j < 12; j++) oacc[j] = vzero8();
  v16h a_cur = load_a_frag(hb, HIDD, lr, 0, lane);
  for (int k0 = 0; k0 < HIDD - 32; k0 += 32) {
    v16h a_nxt = load_a_frag(hb, HIDD, lr, k0 + 32, lane);
#pragma unroll
    for (int j = 0; j < 12; j++)
      oacc[j] = wmma32(a_cur, load_b_frag(W2T, HIDD, j * 16 + lr, k0, lane), oacc[j]);
    a_cur = a_nxt;
  }
#pragma unroll
  for (int j = 0; j < 12; j++)
    oacc[j] = wmma32(a_cur, load_b_frag(W2T, HIDD, j * 16 + lr, HIDD - 32, lane), oacc[j]);

#pragma unroll
  for (int j = 0; j < 12; j++) {
    int n = j * 16 + lr;
    float bv = b2[n];
#pragma unroll
    for (int v = 0; v < 8; v++) {
      size_t m = (size_t)row0 + v + hi * 8;
      out[m * CDIM + n] += oacc[j][v] + bv;     // residual add in place
    }
  }
}

// ---------------------------------------------------------------------------
extern "C" void kernel_launch(void* const* d_in, const int* in_sizes, int n_in,
                              void* d_out, int out_size, void* d_ws, size_t ws_size,
                              hipStream_t stream) {
  (void)in_sizes; (void)n_in; (void)out_size; (void)ws_size;
  const float* se     = (const float*)d_in[0];
  const float* de     = (const float*)d_in[1];
  const float* co     = (const float*)d_in[2];
  const float* maskm  = (const float*)d_in[3];
  const float* n1se_g = (const float*)d_in[4];
  const float* n1se_b = (const float*)d_in[5];
  const float* n1de_g = (const float*)d_in[6];
  const float* n1de_b = (const float*)d_in[7];
  const float* rpb    = (const float*)d_in[8];
  const float* vse_w  = (const float*)d_in[9];
  const float* vse_b  = (const float*)d_in[10];
  const float* vde_w  = (const float*)d_in[11];
  const float* vde_b  = (const float*)d_in[12];
  const float* qk_w   = (const float*)d_in[13];
  const float* qk_b   = (const float*)d_in[14];
  const float* pse_w  = (const float*)d_in[15];
  const float* pse_b  = (const float*)d_in[16];
  const float* pde_w  = (const float*)d_in[17];
  const float* pde_b  = (const float*)d_in[18];
  const float* n2se_g = (const float*)d_in[19];
  const float* n2se_b = (const float*)d_in[20];
  const float* n2de_g = (const float*)d_in[21];
  const float* n2de_b = (const float*)d_in[22];
  const float* mse_w1 = (const float*)d_in[23];
  const float* mse_b1 = (const float*)d_in[24];
  const float* mse_w2 = (const float*)d_in[25];
  const float* mse_b2 = (const float*)d_in[26];
  const float* mde_w1 = (const float*)d_in[27];
  const float* mde_b1 = (const float*)d_in[28];
  const float* mde_w2 = (const float*)d_in[29];
  const float* mde_b2 = (const float*)d_in[30];

  float* out_se = (float*)d_out;
  float* out_de = out_se + (size_t)ROWS * CDIM;

  char* ws = (char*)d_ws;
  size_t off = 0;
  auto alloc = [&](size_t bytes) -> char* {
    char* p = ws + off;
    off += (bytes + 255) & ~(size_t)255;
    return p;
  };
  const size_t RC = (size_t)ROWS * CDIM;
  half_t* vseT  = (half_t*)alloc(384 * 192 * sizeof(half_t));
  half_t* vdeT  = (half_t*)alloc(384 * 192 * sizeof(half_t));
  half_t* qkT   = (half_t*)alloc(384 * 192 * sizeof(half_t));
  half_t* pseT  = (half_t*)alloc(192 * 192 * sizeof(half_t));
  half_t* pdeT  = (half_t*)alloc(192 * 192 * sizeof(half_t));
  half_t* m1seT = (half_t*)alloc(768 * 192 * sizeof(half_t));
  half_t* m2seT = (half_t*)alloc(192 * 768 * sizeof(half_t));
  half_t* m1deT = (half_t*)alloc(768 * 192 * sizeof(half_t));
  half_t* m2deT = (half_t*)alloc(192 * 768 * sizeof(half_t));
  half_t* xw_se = (half_t*)alloc(RC * sizeof(half_t));
  half_t* xw_de = (half_t*)alloc(RC * sizeof(half_t));
  half_t* xw_co = (half_t*)alloc(RC * sizeof(half_t));
  half_t* v_se  = (half_t*)alloc(RC * sizeof(half_t));
  half_t* v_de  = (half_t*)alloc(RC * sizeof(half_t));
  half_t* qkbuf = (half_t*)alloc((size_t)ROWS * 384 * sizeof(half_t));
  half_t* att_se = (half_t*)alloc(RC * sizeof(half_t));
  half_t* att_de = (half_t*)alloc(RC * sizeof(half_t));
  float*  proj_se = (float*)alloc(RC * sizeof(float));
  float*  proj_de = (float*)alloc(RC * sizeof(float));
  // qkbuf is dead after attention -> reuse it for the LN2 f16 outputs.
  half_t* x2_se = qkbuf;
  half_t* x2_de = qkbuf + RC;

  auto wgrid = [](int n) { return (n + 255) / 256; };
  wt_f16<<<wgrid(384 * 192), 256, 0, stream>>>(vse_w, vseT, 384, 192);
  wt_f16<<<wgrid(384 * 192), 256, 0, stream>>>(vde_w, vdeT, 384, 192);
  wt_f16<<<wgrid(192 * 384), 256, 0, stream>>>(qk_w,  qkT,  192, 384);
  wt_f16<<<wgrid(192 * 192), 256, 0, stream>>>(pse_w, pseT, 192, 192);
  wt_f16<<<wgrid(192 * 192), 256, 0, stream>>>(pde_w, pdeT, 192, 192);
  wt_f16<<<wgrid(192 * 768), 256, 0, stream>>>(mse_w1, m1seT, 192, 768);
  wt_f16<<<wgrid(768 * 192), 256, 0, stream>>>(mse_w2, m2seT, 768, 192);
  wt_f16<<<wgrid(192 * 768), 256, 0, stream>>>(mde_w1, m1deT, 192, 768);
  wt_f16<<<wgrid(768 * 192), 256, 0, stream>>>(mde_w2, m2deT, 768, 192);

  ln_shift_part<<<ROWS / 8, 256, 0, stream>>>(se, de, co, n1se_g, n1se_b, n1de_g, n1de_b,
                                              xw_se, xw_de, xw_co);

  dim3 g3(ROWS / 16 / 8, 3), g6(ROWS / 16 / 8, 6);
  // V projections: concat([x, co]) @ W == x @ W[:192] + co @ W[192:]
  gemm_f16w<384, true,  true ><<<g3, 256, 0, stream>>>(xw_se, xw_co, vseT, vse_b, v_se, 192);
  gemm_f16w<384, true,  true ><<<g3, 256, 0, stream>>>(xw_de, xw_co, vdeT, vde_b, v_de, 192);
  gemm_f16w<192, false, true ><<<g6, 256, 0, stream>>>(xw_co, nullptr, qkT, qk_b, qkbuf, 384);

  attn_win<<<dim3(1024, 6), 32, 0, stream>>>(qkbuf, v_se, v_de, rpb, maskm, att_se, att_de);

  gemm_f16w<192, false, false><<<g3, 256, 0, stream>>>(att_se, nullptr, pseT, pse_b, proj_se, 192);
  gemm_f16w<192, false, false><<<g3, 256, 0, stream>>>(att_de, nullptr, pdeT, pde_b, proj_de, 192);

  scatter_residual<<<(ROWS * CDIM) / 256, 256, 0, stream>>>(se, de, proj_se, proj_de,
                                                            out_se, out_de);

  ln_rows<<<ROWS / 8, 256, 0, stream>>>(out_se, n2se_g, n2se_b, x2_se);
  ln_rows<<<ROWS / 8, 256, 0, stream>>>(out_de, n2de_g, n2de_b, x2_de);

  mlp_fused<<<ROWS / 16 / MLPW, 64, 0, stream>>>(x2_se, m1seT, mse_b1, m2seT, mse_b2, out_se);
  mlp_fused<<<ROWS / 16 / MLPW, 64, 0, stream>>>(x2_de, m1deT, mde_b1, m2deT, mde_b2, out_de);
}
